// Summary_encoder_2559800509088
// MI455X (gfx1250) — matrix operand order repaired and checked by reading, not compiled, not run on hardware
//
#include <hip/hip_runtime.h>

// Problem constants (from reference): V=32000, E=H=1024, L=2, B=64, T=512
#define E_   1024
#define H_   1024
#define L_   2
#define B_   64
#define T_   512
#define G3   3072      // 3*H
#define CHUNK 64       // timesteps of gx precomputed per big GEMM
#define BK    32       // K-slab per LDS tile (one WMMA K step)
#define LDSTR 40       // LDS row stride in bf16 elems (BK + 8 pad -> 80B stride)

typedef __attribute__((ext_vector_type(16))) __bf16 v16bf;
typedef __attribute__((ext_vector_type(8)))  __bf16 v8bf;
typedef __attribute__((ext_vector_type(8)))  float  v8f;
typedef __attribute__((ext_vector_type(4)))  int    v4i;

// ---- CDNA5 async global->LDS path (ASYNCcnt), guarded so we always compile ----
#if defined(__AMDGCN__) && __has_builtin(__builtin_amdgcn_global_load_async_to_lds_b128) && \
    __has_builtin(__builtin_amdgcn_s_wait_asynccnt)
#define USE_ASYNC 1
#else
#define USE_ASYNC 0
#endif

#if USE_ASYNC
// Builtin signature (from compiler diagnostic): arg0 = AS1 v4i*, arg1 = AS3 v4i*.
typedef __attribute__((address_space(1))) v4i gv4i;
typedef __attribute__((address_space(3))) v4i lv4i;
static __device__ __forceinline__ gv4i* to_g(const void* p) {
    return (gv4i*)(uintptr_t)p;              // generic global addr == AS1 addr
}
static __device__ __forceinline__ lv4i* to_l(void* p) {
    return (lv4i*)(uint32_t)(uintptr_t)p;    // low 32 bits of generic LDS addr = LDS offset
}
#define WAIT_ASYNC(n) __builtin_amdgcn_s_wait_asynccnt(n)
#else
#define WAIT_ASYNC(n)
#endif

// Copy 16 bf16 (32B) global -> LDS as two b128 moves (async when available).
static __device__ __forceinline__ void cp_chunk16(const __bf16* g, __bf16* l) {
#if USE_ASYNC
    __builtin_amdgcn_global_load_async_to_lds_b128(to_g(g), to_l(l), 0, 0);
    __builtin_amdgcn_global_load_async_to_lds_b128(to_g(g + 8), to_l(l + 8), 0, 0);
#else
    *reinterpret_cast<v8bf*>(l)     = *reinterpret_cast<const v8bf*>(g);
    *reinterpret_cast<v8bf*>(l + 8) = *reinterpret_cast<const v8bf*>(g + 8);
#endif
}

static __device__ __forceinline__ v8f vzero8() {
    v8f z;
#pragma unroll
    for (int i = 0; i < 8; ++i) z[i] = 0.0f;
    return z;
}

// A-operand fragment (16x32 bf16): lanes 0-15 hold K {k0..k0+7, k0+16..k0+23},
// lanes 16-31 shifted by 8 (ISA 7.12.2). rowp points at this lane's M-row, k col 0.
static __device__ __forceinline__ v16bf a_frag(const __bf16* rowp, int lhalf) {
    union { v16bf v; v8bf h[2]; } u;
    u.h[0] = *reinterpret_cast<const v8bf*>(rowp + lhalf * 8);
    u.h[1] = *reinterpret_cast<const v8bf*>(rowp + lhalf * 8 + 16);
    return u.v;
}
// B-operand fragment (32x16 bf16) from a row-major W[N][K] row (== B^T):
// lane n=lane&15, K {k0..k0+7, k0+16..k0+23} shifted by (lane>>4)*8 (same as A).
static __device__ __forceinline__ v16bf b_frag(const __bf16* rowp, int lhalf) {
    union { v16bf v; v8bf h[2]; } u;
    u.h[0] = *reinterpret_cast<const v8bf*>(rowp + lhalf * 8);
    u.h[1] = *reinterpret_cast<const v8bf*>(rowp + lhalf * 8 + 16);
    return u.v;
}

static __device__ __forceinline__ v8f wmma_bf16(v16bf a, v16bf b, v8f c) {
    return __builtin_amdgcn_wmma_f32_16x16x32_bf16(false, a, false, b, (short)0, c,
                                                   false, false);
}

static __device__ __forceinline__ float sigmoid_f(float x) {
    return 1.0f / (1.0f + __expf(-x));
}
static __device__ __forceinline__ float tanh_f(float x) {
    float e2 = __expf(-2.0f * x);
    return (1.0f - e2) / (1.0f + e2);
}

// ---------------------------------------------------------------------------
// Elementwise helpers
// ---------------------------------------------------------------------------
__global__ void f32_to_bf16_kernel(const float* __restrict__ src,
                                   __bf16* __restrict__ dst, int n) {
    int i = blockIdx.x * blockDim.x + threadIdx.x;
    if (i < n) dst[i] = (__bf16)src[i];
}

// xs[t][b][e] = embedding[x[b][t]][e]  (bf16)
__global__ void embed_kernel(const int* __restrict__ x,
                             const float* __restrict__ emb,
                             __bf16* __restrict__ xs) {
    size_t i = (size_t)blockIdx.x * blockDim.x + threadIdx.x;   // over T*B*E
    int e  = (int)(i % E_);
    int tb = (int)(i / E_);
    int b  = tb % B_;
    int t  = tb / B_;
    int tok = x[b * T_ + t];
    xs[i] = (__bf16)emb[(size_t)tok * E_ + e];
}

__global__ void init_h_kernel(const float* __restrict__ h0l,
                              float* __restrict__ hf, __bf16* __restrict__ hbf) {
    int i = blockIdx.x * blockDim.x + threadIdx.x;
    if (i < B_ * H_) {
        float v = h0l[i];
        hf[i]  = v;
        hbf[i] = (__bf16)v;
    }
}

__global__ void copy_f32_kernel(const float* __restrict__ s,
                                float* __restrict__ d, int n) {
    int i = blockIdx.x * blockDim.x + threadIdx.x;
    if (i < n) d[i] = s[i];
}

// ---------------------------------------------------------------------------
// Big input-side GEMM: C[M,3072] = A[M,1024] @ W[3072,1024]^T + bias
// 256 threads (8 waves); block tile 128x128; wave tile 32x64.
// A and B slabs double-buffered in LDS; fills use async-to-LDS, waited with
// s_wait_asynccnt so the next slab streams in under the current WMMAs.
// ---------------------------------------------------------------------------
__global__ __launch_bounds__(256) void gemm_gx_kernel(
    const __bf16* __restrict__ A,    // [M, 1024]
    const __bf16* __restrict__ W,    // [3072, 1024]
    const float* __restrict__ bias,  // [3072]
    float* __restrict__ C,           // [M, 3072]
    int M) {
    __shared__ __bf16 As[2][128 * LDSTR];
    __shared__ __bf16 Bs[2][128 * LDSTR];
    const int K = 1024, N = G3;

    int tid   = threadIdx.x;
    int lane  = tid & 31;
    int wave  = tid >> 5;          // 0..7
    int l15   = lane & 15;
    int lhalf = lane >> 4;
    int m_blk = blockIdx.y * 128;
    int n_blk = blockIdx.x * 128;
    int wm    = (wave & 3) * 32;   // 4 waves along M
    int wn    = (wave >> 2) * 64;  // 2 waves along N

    // fill mapping: thread -> one 32B chunk of A slab + one of B slab
    int frow = tid >> 1;           // 0..127
    int fcol = (tid & 1) * 16;     // 0 or 16
    const __bf16* gA = A + (size_t)(m_blk + frow) * K + fcol;
    const __bf16* gB = W + (size_t)(n_blk + frow) * K + fcol;

    v8f acc[2][4];
#pragma unroll
    for (int mt = 0; mt < 2; ++mt)
#pragma unroll
        for (int nt = 0; nt < 4; ++nt) acc[mt][nt] = vzero8();

    // prologue fill of buffer 0
    cp_chunk16(gA, &As[0][frow * LDSTR + fcol]);
    cp_chunk16(gB, &Bs[0][frow * LDSTR + fcol]);

    for (int k0 = 0; k0 < K; k0 += BK) {
        int buf = (k0 >> 5) & 1;
        if (k0 + BK < K) {  // prefetch next slab, leave its 4 ops in flight
            cp_chunk16(gA + k0 + BK, &As[buf ^ 1][frow * LDSTR + fcol]);
            cp_chunk16(gB + k0 + BK, &Bs[buf ^ 1][frow * LDSTR + fcol]);
            WAIT_ASYNC(4);
        } else {
            WAIT_ASYNC(0);
        }
        __syncthreads();

        v16bf a0 = a_frag(&As[buf][(wm + l15) * LDSTR], lhalf);
        v16bf a1 = a_frag(&As[buf][(wm + 16 + l15) * LDSTR], lhalf);
#pragma unroll
        for (int nt = 0; nt < 4; ++nt) {
            v16bf b = b_frag(&Bs[buf][(wn + nt * 16 + l15) * LDSTR], lhalf);
            acc[0][nt] = wmma_bf16(a0, b, acc[0][nt]);
            acc[1][nt] = wmma_bf16(a1, b, acc[1][nt]);
        }
        __syncthreads();   // slab consumed; safe to overwrite next iteration
    }

#pragma unroll
    for (int mt = 0; mt < 2; ++mt)
#pragma unroll
        for (int nt = 0; nt < 4; ++nt) {
            int col  = n_blk + wn + nt * 16 + l15;
            float bv = bias[col];
#pragma unroll
            for (int i = 0; i < 8; ++i) {
                int row = m_blk + wm + mt * 16 + lhalf * 8 + i;
                C[(size_t)row * N + col] = acc[mt][nt][i] + bv;
            }
        }
}

// ---------------------------------------------------------------------------
// One GRU timestep: gh = h_prev @ Whh^T with fused gate math.
// grid = 16 blocks (64 cols of H each), 128 threads (4 waves = 4 x 16 batch rows).
// The 192x32 Whh slab (3 gates x 64 cols) is shared by all 4 waves ->
// double-buffered in LDS via async-to-LDS. h_prev rows stay direct global
// loads (no cross-wave redundancy, hot in WGP$).
// ---------------------------------------------------------------------------
__global__ __launch_bounds__(128) void gru_step_kernel(
    const __bf16* __restrict__ hprev_bf,  // [B, H]
    const float* __restrict__ hprev_f,    // [B, H]
    const __bf16* __restrict__ Whh,       // [3H, H]
    const float* __restrict__ bhh,        // [3H]
    const float* __restrict__ gx_t,       // [B, 3H]  (includes b_ih)
    float* __restrict__ hnew_f,           // [B, H]
    __bf16* __restrict__ hnew_bf,         // [B, H]
    float* __restrict__ hiddens,          // [B, T, H] or nullptr
    int t,
    __bf16* __restrict__ ys_t)            // [B, H] slice or nullptr
{
    __shared__ __bf16 Bs[2][192 * LDSTR];

    int tid   = threadIdx.x;
    int lane  = tid & 31;
    int wave  = tid >> 5;   // 0..3 -> batch row groups
    int l15   = lane & 15;
    int lhalf = lane >> 4;
    int m0    = wave * 16;
    int jbase = blockIdx.x * 64;

    // fill mapping: 192 rows x 2 halves = 384 chunks; 3 chunks per thread.
    // local row rl in [0,192): global W row = (rl>>6)*H + jbase + (rl&63)
    const __bf16* gsrc[3];
    int lrow[3], lcol[3];
#pragma unroll
    for (int j = 0; j < 3; ++j) {
        int cid = tid + j * 128;
        int rl  = cid >> 1;
        lrow[j] = rl;
        lcol[j] = (cid & 1) * 16;
        gsrc[j] = Whh + (size_t)((rl >> 6) * H_ + jbase + (rl & 63)) * H_ + lcol[j];
    }

    v8f acc[3][4];
#pragma unroll
    for (int g = 0; g < 3; ++g)
#pragma unroll
        for (int nt = 0; nt < 4; ++nt) acc[g][nt] = vzero8();

    const __bf16* arow = hprev_bf + (size_t)(m0 + l15) * H_;

#pragma unroll
    for (int j = 0; j < 3; ++j)
        cp_chunk16(gsrc[j], &Bs[0][lrow[j] * LDSTR + lcol[j]]);

    for (int k0 = 0; k0 < H_; k0 += BK) {
        int buf = (k0 >> 5) & 1;
        if (k0 + BK < H_) {
#pragma unroll
            for (int j = 0; j < 3; ++j)
                cp_chunk16(gsrc[j] + k0 + BK, &Bs[buf ^ 1][lrow[j] * LDSTR + lcol[j]]);
            WAIT_ASYNC(6);
        } else {
            WAIT_ASYNC(0);
        }
        __syncthreads();

        v16bf a = a_frag(arow + k0, lhalf);
#pragma unroll
        for (int g = 0; g < 3; ++g)
#pragma unroll
            for (int nt = 0; nt < 4; ++nt) {
                v16bf b = b_frag(&Bs[buf][(g * 64 + nt * 16 + l15) * LDSTR], lhalf);
                acc[g][nt] = wmma_bf16(a, b, acc[g][nt]);
            }
        __syncthreads();
    }

#pragma unroll
    for (int nt = 0; nt < 4; ++nt) {
        int col  = jbase + nt * 16 + l15;
        float br = bhh[col];
        float bz = bhh[H_ + col];
        float bn = bhh[2 * H_ + col];
#pragma unroll
        for (int i = 0; i < 8; ++i) {
            int b = m0 + lhalf * 8 + i;
            const float* gxb = gx_t + (size_t)b * G3;
            float ghr = acc[0][nt][i] + br;
            float ghz = acc[1][nt][i] + bz;
            float ghn = acc[2][nt][i] + bn;
            float r = sigmoid_f(gxb[col] + ghr);
            float z = sigmoid_f(gxb[H_ + col] + ghz);
            float n = tanh_f(gxb[2 * H_ + col] + r * ghn);
            size_t hi = (size_t)b * H_ + col;
            float hv = (1.0f - z) * n + z * hprev_f[hi];
            hnew_f[hi]  = hv;
            hnew_bf[hi] = (__bf16)hv;
            if (hiddens)
                hiddens[(size_t)b * ((size_t)T_ * H_) + (size_t)t * H_ + col] = hv;
            if (ys_t) ys_t[hi] = (__bf16)hv;
        }
    }
}

// ---------------------------------------------------------------------------
// Host orchestration (graph-capture safe: only kernel launches on `stream`)
// ---------------------------------------------------------------------------
extern "C" void kernel_launch(void* const* d_in, const int* in_sizes, int n_in,
                              void* d_out, int out_size, void* d_ws, size_t ws_size,
                              hipStream_t stream) {
    const int*   x   = (const int*)d_in[0];
    const float* h0  = (const float*)d_in[1];
    const float* emb = (const float*)d_in[2];
    const float* Wih = (const float*)d_in[3];
    const float* Whh = (const float*)d_in[4];
    const float* bih = (const float*)d_in[5];
    const float* bhh = (const float*)d_in[6];
    float* out = (float*)d_out;

    char* ws = (char*)d_ws;
    size_t off = 0;
    auto alloc = [&](size_t bytes) -> void* {
        void* p = ws + off;
        off += (bytes + 255) & ~(size_t)255;
        return p;
    };
    __bf16* wih_bf = (__bf16*)alloc((size_t)L_ * G3 * E_ * 2);
    __bf16* whh_bf = (__bf16*)alloc((size_t)L_ * G3 * H_ * 2);
    __bf16* xs_bf  = (__bf16*)alloc((size_t)T_ * B_ * E_ * 2);   // layer0 input [T,B,E]
    __bf16* ys_bf  = (__bf16*)alloc((size_t)T_ * B_ * H_ * 2);   // layer0 output [T,B,H]
    float*  gx     = (float*)alloc((size_t)CHUNK * B_ * G3 * 4); // chunked x-side gates
    float*  hAf    = (float*)alloc((size_t)B_ * H_ * 4);
    float*  hBf    = (float*)alloc((size_t)B_ * H_ * 4);
    __bf16* hAbf   = (__bf16*)alloc((size_t)B_ * H_ * 2);
    __bf16* hBbf   = (__bf16*)alloc((size_t)B_ * H_ * 2);

    // Convert weights to bf16 once.
    int nW = L_ * G3 * E_;
    f32_to_bf16_kernel<<<(nW + 255) / 256, 256, 0, stream>>>(Wih, wih_bf, nW);
    f32_to_bf16_kernel<<<(nW + 255) / 256, 256, 0, stream>>>(Whh, whh_bf, nW);

    // Embedding gather + bf16 convert: xs[t][b][e]
    size_t nxs = (size_t)T_ * B_ * E_;
    embed_kernel<<<(unsigned)(nxs / 256), 256, 0, stream>>>(x, emb, xs_bf);

    float* out_out     = out;                                 // [1,B,H]
    float* out_hidden  = out + (size_t)B_ * H_;               // [L,B,H]
    float* out_hiddens = out_hidden + (size_t)L_ * B_ * H_;   // [B,T,H]

    for (int l = 0; l < L_; ++l) {
        const __bf16* layer_in = (l == 0) ? xs_bf : ys_bf;    // [T,B,1024]
        const __bf16* wih_l = wih_bf + (size_t)l * G3 * E_;
        const __bf16* whh_l = whh_bf + (size_t)l * G3 * H_;
        const float*  bih_l = bih + l * G3;
        const float*  bhh_l = bhh + l * G3;

        init_h_kernel<<<(B_ * H_ + 255) / 256, 256, 0, stream>>>(
            h0 + (size_t)l * B_ * H_, hAf, hAbf);

        int flip = 0;
        for (int c0 = 0; c0 < T_; c0 += CHUNK) {
            // gx[chunk] = layer_in[chunk] @ Wih^T + bih   (M = CHUNK*B = 4096)
            dim3 ggrid(G3 / 128, (CHUNK * B_) / 128);
            gemm_gx_kernel<<<ggrid, 256, 0, stream>>>(
                layer_in + (size_t)c0 * B_ * E_, wih_l, bih_l, gx, CHUNK * B_);

            for (int t = c0; t < c0 + CHUNK; ++t) {
                const float*  hpf = flip ? hBf : hAf;
                const __bf16* hpb = flip ? hBbf : hAbf;
                float*  hnf = flip ? hAf : hBf;
                __bf16* hnb = flip ? hAbf : hBbf;
                gru_step_kernel<<<H_ / 64, 128, 0, stream>>>(
                    hpb, hpf, whh_l, bhh_l, gx + (size_t)(t - c0) * B_ * G3,
                    hnf, hnb,
                    (l == 0) ? out_hiddens : (float*)nullptr, t,
                    (l == 0) ? (ys_bf + (size_t)t * B_ * H_) : (__bf16*)nullptr);
                flip ^= 1;
            }
        }
        // T even -> final hidden ended in hAf (flip wrapped back to 0)
        const float* hfin = flip ? hBf : hAf;
        copy_f32_kernel<<<(B_ * H_ + 255) / 256, 256, 0, stream>>>(
            hfin, out_hidden + (size_t)l * B_ * H_, B_ * H_);
        if (l == L_ - 1)
            copy_f32_kernel<<<(B_ * H_ + 255) / 256, 256, 0, stream>>>(
                hfin, out_out, B_ * H_);
    }
}
